// DifferentialAttention_72060961292809
// MI455X (gfx1250) — matrix operand order repaired and checked by reading, not compiled, not run on hardware
//
#include <hip/hip_runtime.h>
#include <math.h>

typedef unsigned short u16;
typedef unsigned int   u32;

typedef __attribute__((ext_vector_type(16))) __bf16 v16bf;
typedef __attribute__((ext_vector_type(8)))  float  v8f;
typedef __attribute__((ext_vector_type(4)))  int    v4i;

union BF16x16 { v16bf v; uint4 q[2]; };

#define AS1 __attribute__((address_space(1)))
#define AS3 __attribute__((address_space(3)))

#if defined(__has_builtin)
#  if __has_builtin(__builtin_amdgcn_global_load_async_to_lds_b128)
#    define HAVE_ASYNC_LDS 1
#  endif
#  if __has_builtin(__builtin_amdgcn_s_wait_asynccnt)
#    define HAVE_WAIT_ASYNC 1
#  endif
#endif

// ---------------------------------------------------------------------------
// helpers
// ---------------------------------------------------------------------------
__device__ inline v8f zero8() {
  v8f z = {0.f,0.f,0.f,0.f,0.f,0.f,0.f,0.f};
  return z;
}

__device__ inline u16 f2bf(float f) {
  u32 u = __float_as_uint(f);
  u32 r = u + 0x7FFFu + ((u >> 16) & 1u);   // round-to-nearest-even
  return (u16)(r >> 16);
}

// A-matrix tile (16x32 bf16).  lane L%16 owns row M; per ISA layout the 16
// elements are K = half*8 + (0..7) and K = 16 + half*8 + (0..7)  (half = L/16)
__device__ inline v16bf ld_a(const u16* __restrict__ row, int k0, int half) {
  BF16x16 t;
  t.q[0] = *(const uint4*)(row + k0 + half * 8);
  t.q[1] = *(const uint4*)(row + k0 + 16 + half * 8);
  return t.v;
}

// B-matrix tile (32x16 bf16).  lane owns column N = L%16; elements are
// K = half*8 + (0..7) and K = 16 + half*8 + (0..7), matching A.
__device__ inline v16bf ld_b(const u16* __restrict__ row, int k0, int half) {
  const u16* p = row + k0 + half * 8;
  BF16x16 t;
  t.q[0] = *(const uint4*)(p);
  t.q[1] = *(const uint4*)(p + 16);
  return t.v;
}

__device__ inline v8f wmma_bf16(v16bf a, v16bf b, v8f c) {
  return __builtin_amdgcn_wmma_f32_16x16x32_bf16(
      /*neg_a=*/false, a, /*neg_b=*/false, b,
      /*c_mod=*/(short)0, c, /*reuse_a=*/false, /*reuse_b=*/false);
}

// async (or sync-fallback) 16B copy global -> LDS
__device__ inline void stage_chunk(const u16* gsrc, u16* lds_dst) {
#if defined(HAVE_ASYNC_LDS)
  __builtin_amdgcn_global_load_async_to_lds_b128(
      (AS1 v4i*)gsrc, (AS3 v4i*)lds_dst, 0, 0);
#else
  *(uint4*)lds_dst = *(const uint4*)gsrc;
#endif
}

__device__ inline void wait_async_4() {
#if defined(HAVE_ASYNC_LDS)
#  if defined(HAVE_WAIT_ASYNC)
  __builtin_amdgcn_s_wait_asynccnt(4);
#  else
  asm volatile("s_wait_asynccnt 4" ::: "memory");
#  endif
#endif
}

__device__ inline void wait_async_0() {
#if defined(HAVE_ASYNC_LDS)
#  if defined(HAVE_WAIT_ASYNC)
  __builtin_amdgcn_s_wait_asynccnt(0);
#  else
  asm volatile("s_wait_asynccnt 0" ::: "memory");
#  endif
#endif
}

// ---------------------------------------------------------------------------
// kernel: lambda_full = exp(sum lq1*lk1) - exp(sum lq2*lk2) + 0.8
// ---------------------------------------------------------------------------
__global__ __launch_bounds__(256) void k_lambda(
    const float* __restrict__ lq1, const float* __restrict__ lk1,
    const float* __restrict__ lq2, const float* __restrict__ lk2,
    float* __restrict__ lam) {
  __shared__ float s1[256], s2[256];
  int i = threadIdx.x;
  s1[i] = lq1[i] * lk1[i];
  s2[i] = lq2[i] * lk2[i];
  __syncthreads();
  for (int st = 128; st; st >>= 1) {
    if (i < st) { s1[i] += s1[i + st]; s2[i] += s2[i + st]; }
    __syncthreads();
  }
  if (i == 0) lam[0] = expf(s1[0]) - expf(s2[0]) + 0.8f;
}

// ---------------------------------------------------------------------------
// kernel: f32 -> bf16 (vectorized x4)
// ---------------------------------------------------------------------------
__global__ __launch_bounds__(256) void k_cvt_bf16(
    const float* __restrict__ in, u16* __restrict__ out, int n4) {
  int i = blockIdx.x * 256 + threadIdx.x;
  if (i >= n4) return;
  float4 f = ((const float4*)in)[i];
  uint2 r;
  r.x = (u32)f2bf(f.x) | ((u32)f2bf(f.y) << 16);
  r.y = (u32)f2bf(f.z) | ((u32)f2bf(f.w) << 16);
  ((uint2*)out)[i] = r;
}

// ---------------------------------------------------------------------------
// kernel: GEMM  C[4096x2048] = A[4096x2048] @ W[2048x2048]^T  * scale
// A, W bf16 row-major (pitch 2048), C f32.
// Block tile 128x128, 8 waves x (32x64) wave tiles.  K-panels (128x32 for A
// and for W) are double-buffered in LDS and filled with async-to-LDS loads;
// s_wait_asynccnt 4 releases the previous panel while the next is in flight.
// LDS rows padded to 80B: 16B-aligned and conflict-free for ds_load_b128.
// ---------------------------------------------------------------------------
__global__ __launch_bounds__(256) void k_gemm_bf16(
    const u16* __restrict__ A, const u16* __restrict__ W,
    float* __restrict__ C, float scale) {
  __shared__ u16 sA[2][128][40];
  __shared__ u16 sB[2][128][40];
  const int t    = threadIdx.x;
  const int lane = t & 31;
  const int w    = t >> 5;
  const int half = lane >> 4;
  const int wrow = w >> 1, wcol = w & 1;           // 4 x 2 wave grid
  const int M0 = blockIdx.y * 128;
  const int N0 = blockIdx.x * 128;

  // each thread stages 2 x 16B chunks per panel per K-step
  const int c0row = t >> 2,           c0k = (t & 3) * 8;
  const int c1row = (t + 256) >> 2,   c1k = (t & 3) * 8;   // (t+256)&3 == t&3

  v8f acc[2][4];
#pragma unroll
  for (int mi = 0; mi < 2; mi++)
#pragma unroll
    for (int j = 0; j < 4; j++) acc[mi][j] = zero8();

  // prologue: stage K-step 0 into buffer 0
  stage_chunk(A + (size_t)(M0 + c0row) * 2048 + c0k, &sA[0][c0row][c0k]);
  stage_chunk(A + (size_t)(M0 + c1row) * 2048 + c1k, &sA[0][c1row][c1k]);
  stage_chunk(W + (size_t)(N0 + c0row) * 2048 + c0k, &sB[0][c0row][c0k]);
  stage_chunk(W + (size_t)(N0 + c1row) * 2048 + c1k, &sB[0][c1row][c1k]);

  for (int i = 0; i < 64; i++) {
    const int cur = i & 1;
    if (i + 1 < 64) {
      const int nk = (i + 1) * 32;
      stage_chunk(A + (size_t)(M0 + c0row) * 2048 + nk + c0k, &sA[cur ^ 1][c0row][c0k]);
      stage_chunk(A + (size_t)(M0 + c1row) * 2048 + nk + c1k, &sA[cur ^ 1][c1row][c1k]);
      stage_chunk(W + (size_t)(N0 + c0row) * 2048 + nk + c0k, &sB[cur ^ 1][c0row][c0k]);
      stage_chunk(W + (size_t)(N0 + c1row) * 2048 + nk + c1k, &sB[cur ^ 1][c1row][c1k]);
      wait_async_4();          // previous panel complete; next still in flight
    } else {
      wait_async_0();
    }
    __syncthreads();           // panel[cur] visible to all waves

    const u16* arow0 = &sA[cur][wrow * 32 + (lane & 15)][0];
    const u16* arow1 = arow0 + 16 * 40;
    v16bf a0 = ld_a(arow0, 0, half);
    v16bf a1 = ld_a(arow1, 0, half);
#pragma unroll
    for (int j = 0; j < 4; j++) {
      const u16* brow = &sB[cur][wcol * 64 + j * 16 + (lane & 15)][0];
      v16bf bb = ld_b(brow, 0, half);
      acc[0][j] = wmma_bf16(a0, bb, acc[0][j]);
      acc[1][j] = wmma_bf16(a1, bb, acc[1][j]);
    }
    __syncthreads();           // all waves done reading panel[cur]
  }

#pragma unroll
  for (int mi = 0; mi < 2; mi++)
#pragma unroll
    for (int j = 0; j < 4; j++)
#pragma unroll
      for (int r = 0; r < 8; r++) {
        int row = M0 + wrow * 32 + mi * 16 + r + half * 8;
        int col = N0 + wcol * 64 + j * 16 + (lane & 15);
        C[(size_t)row * 2048 + col] = acc[mi][j][r] * scale;
      }
}

// ---------------------------------------------------------------------------
// kernel: RMS-norm every contiguous 128-chunk, f32 in -> bf16 out
// one wave per chunk
// ---------------------------------------------------------------------------
__global__ __launch_bounds__(256) void k_rmsnorm128(
    const float* __restrict__ in, u16* __restrict__ out) {
  int wave = (blockIdx.x * 256 + threadIdx.x) >> 5;
  int lane = threadIdx.x & 31;
  size_t base = (size_t)wave * 128 + lane * 4;
  float4 f = *(const float4*)(in + base);
  float ss = f.x * f.x + f.y * f.y + f.z * f.z + f.w * f.w;
#pragma unroll
  for (int m = 16; m; m >>= 1) ss += __shfl_xor(ss, m, 32);
  float sc = rsqrtf(ss * (1.0f / 128.0f) + 1.1920929e-07f);
  uint2 r;
  r.x = (u32)f2bf(f.x * sc) | ((u32)f2bf(f.y * sc) << 16);
  r.y = (u32)f2bf(f.z * sc) | ((u32)f2bf(f.w * sc) << 16);
  *(uint2*)(out + base) = r;
}

// ---------------------------------------------------------------------------
// kernel: V[b,t,h*256+d] (f32) -> Vt[(b*8+h)*256+d][t] (bf16)
// ---------------------------------------------------------------------------
__global__ __launch_bounds__(256) void k_transpose_v(
    const float* __restrict__ v, u16* __restrict__ vt) {
  int i  = blockIdx.x * 256 + threadIdx.x;   // total 2*8*256*2048
  int t  = i & 2047;
  int d  = (i >> 11) & 255;
  int bh = i >> 19;
  int b = bh >> 3, h = bh & 7;
  float val = v[((size_t)(b * 2048 + t)) * 2048 + h * 256 + d];
  vt[i] = f2bf(val);
}

// ---------------------------------------------------------------------------
// attention pass A: per (b,h,stream,16-query tile) wave, compute row sums
// l = sum_{kv<=q} exp(score).  RMS-norm bounds |score|<=sqrt(128) -> no max
// subtraction needed in f32.
// ---------------------------------------------------------------------------
__global__ __launch_bounds__(256) void k_attn_sumexp(
    const u16* __restrict__ Qn, const u16* __restrict__ Kn,
    float* __restrict__ lsum) {
  const int lane = threadIdx.x & 31, w = threadIdx.x >> 5;
  const int W = blockIdx.x * 8 + w;
  const int qt = W & 127, s = (W >> 7) & 1, h = (W >> 8) & 7, b = W >> 11;
  const int qbase = qt << 4;
  const int colbase = h * 256 + s * 128;
  const int half = lane >> 4;
  const float sc = 0.08838834764831845f;    // 1/sqrt(128)

  const u16* qrow = Qn + ((size_t)(b * 2048 + qbase + (lane & 15))) * 2048;
  v16bf aq[4];
#pragma unroll
  for (int c = 0; c < 4; c++) aq[c] = ld_a(qrow, colbase + c * 32, half);

  float ls[8];
#pragma unroll
  for (int r = 0; r < 8; r++) ls[r] = 0.0f;

  const int nkv = (qbase >> 4) + 1;
  for (int kb = 0; kb < nkv; kb++) {
    int kv0 = kb << 4;
    const u16* krow = Kn + ((size_t)(b * 2048 + kv0 + (lane & 15))) * 2048;
    v8f c = zero8();
#pragma unroll
    for (int cc = 0; cc < 4; cc++)
      c = wmma_bf16(aq[cc], ld_b(krow, colbase + cc * 32, half), c);
    int kv_g = kv0 + (lane & 15);
#pragma unroll
    for (int r = 0; r < 8; r++) {
      int q_g = qbase + r + half * 8;
      float e = (kv_g <= q_g) ? __expf(c[r] * sc) : 0.0f;
      ls[r] += e;
    }
  }
#pragma unroll
  for (int r = 0; r < 8; r++) {
    float v = ls[r];
#pragma unroll
    for (int m = 8; m; m >>= 1) v += __shfl_xor(v, m, 16);
    if ((lane & 15) == 0) {
      int q_g = qbase + r + half * 8;
      lsum[((size_t)(s * 16 + b * 8 + h)) * 2048 + q_g] = v;
    }
  }
}

// ---------------------------------------------------------------------------
// attention pass B: per (b,h,16-query tile, 128-dim chunk) wave.
// Combined prob tile P = exp(s1)/l1 - lam*exp(s2)/l2 built through a per-wave
// LDS tile (C-layout -> A-layout), then Y += P @ V via WMMA.
// ---------------------------------------------------------------------------
__global__ __launch_bounds__(256) void k_attn_out(
    const u16* __restrict__ Qn, const u16* __restrict__ Kn,
    const u16* __restrict__ Vt, const float* __restrict__ lsum,
    const float* __restrict__ lamp, float* __restrict__ Y) {
  __shared__ u16 pt[8][16][32];              // per-wave 16x32 bf16 P tile
  const int lane = threadIdx.x & 31, w = threadIdx.x >> 5;
  const int W = blockIdx.x * 8 + w;
  const int dc = W & 1, qt = (W >> 1) & 127, h = (W >> 8) & 7, b = W >> 11;
  const int qbase = qt << 4, dbase = dc * 128;
  const int half = lane >> 4;
  const float sc = 0.08838834764831845f;
  const float lam = lamp[0];

  float inv1[8], inv2[8];
#pragma unroll
  for (int r = 0; r < 8; r++) {
    int q_g = qbase + r + half * 8;
    inv1[r] = 1.0f / lsum[((size_t)(b * 8 + h)) * 2048 + q_g];
    inv2[r] = 1.0f / lsum[((size_t)(16 + b * 8 + h)) * 2048 + q_g];
  }

  const u16* qrow = Qn + ((size_t)(b * 2048 + qbase + (lane & 15))) * 2048;
  v16bf aq1[4], aq2[4];
#pragma unroll
  for (int c = 0; c < 4; c++) {
    aq1[c] = ld_a(qrow, h * 256 + c * 32, half);
    aq2[c] = ld_a(qrow, h * 256 + 128 + c * 32, half);
  }

  v8f acc[8];
#pragma unroll
  for (int j = 0; j < 8; j++) acc[j] = zero8();

  const int nkv = (qbase >> 5) + 1;          // kv blocks of 32
  for (int kb = 0; kb < nkv; kb++) {
    int kv0 = kb << 5;
#pragma unroll
    for (int sub = 0; sub < 2; sub++) {
      int kvs = kv0 + sub * 16;
      const u16* krow = Kn + ((size_t)(b * 2048 + kvs + (lane & 15))) * 2048;
      v8f c1 = zero8(), c2 = zero8();
#pragma unroll
      for (int cc = 0; cc < 4; cc++) {
        c1 = wmma_bf16(aq1[cc], ld_b(krow, h * 256 + cc * 32, half), c1);
        c2 = wmma_bf16(aq2[cc], ld_b(krow, h * 256 + 128 + cc * 32, half), c2);
      }
      int kv_g = kvs + (lane & 15);
#pragma unroll
      for (int r = 0; r < 8; r++) {
        int q_g = qbase + r + half * 8;
        float p = 0.0f;
        if (kv_g <= q_g)
          p = __expf(c1[r] * sc) * inv1[r] - lam * __expf(c2[r] * sc) * inv2[r];
        pt[w][r + half * 8][sub * 16 + (lane & 15)] = f2bf(p);
      }
    }
    // LDS unit is in-order within a wave; fence for safety (CDNA5 split cnt)
    asm volatile("s_wait_dscnt 0" ::: "memory");

    v16bf ap;
    {
      const u16* prow = &pt[w][lane & 15][0];
      BF16x16 t;
      t.q[0] = *(const uint4*)(prow + half * 8);
      t.q[1] = *(const uint4*)(prow + 16 + half * 8);
      ap = t.v;
    }
#pragma unroll
    for (int j = 0; j < 8; j++) {
      const u16* vrow =
          Vt + ((size_t)((b * 8 + h) * 256 + dbase + j * 16 + (lane & 15))) * 2048;
      acc[j] = wmma_bf16(ap, ld_b(vrow, kv0, half), acc[j]);
    }
  }

#pragma unroll
  for (int j = 0; j < 8; j++)
#pragma unroll
    for (int r = 0; r < 8; r++) {
      int row = b * 2048 + qbase + r + half * 8;
      int col = h * 256 + dbase + j * 16 + (lane & 15);
      Y[(size_t)row * 2048 + col] = acc[j][r];
    }
}

// ---------------------------------------------------------------------------
// host launcher
// ---------------------------------------------------------------------------
extern "C" void kernel_launch(void* const* d_in, const int* in_sizes, int n_in,
                              void* d_out, int out_size, void* d_ws, size_t ws_size,
                              hipStream_t stream) {
  const float* x   = (const float*)d_in[0];
  const float* wq  = (const float*)d_in[1];
  const float* wk  = (const float*)d_in[2];
  const float* wv  = (const float*)d_in[3];
  const float* wo  = (const float*)d_in[4];
  const float* lq1 = (const float*)d_in[5];
  const float* lk1 = (const float*)d_in[6];
  const float* lq2 = (const float*)d_in[7];
  const float* lk2 = (const float*)d_in[8];

  const size_t MB = 1ull << 20;
  char* ws = (char*)d_ws;
  u16*   XB  = (u16*)(ws + 0);            // x bf16                 16 MB
  u16*   WQB = (u16*)(ws + 16 * MB);      // wq bf16                 8 MB
  u16*   WKB = (u16*)(ws + 24 * MB);      // wk bf16                 8 MB
  u16*   WVB = (u16*)(ws + 32 * MB);      // wv bf16                 8 MB
  u16*   WOB = (u16*)(ws + 40 * MB);      // wo bf16                 8 MB
  float* QF  = (float*)(ws + 48 * MB);    // q f32 (reused as y f32) 32 MB
  float* KF  = (float*)(ws + 80 * MB);    // k f32 (reused as y bf16)32 MB
  float* VF  = (float*)(ws + 112 * MB);   // v f32                   32 MB
  u16*   QN  = (u16*)(ws + 144 * MB);     // q normed bf16           16 MB
  u16*   KN  = (u16*)(ws + 160 * MB);     // k normed bf16           16 MB
  u16*   VT  = (u16*)(ws + 176 * MB);     // v transposed bf16       16 MB
  float* LAM = (float*)(ws + 192 * MB);   // lambda_full
  float* LS  = (float*)(ws + 192 * MB + 256); // row sums 2*16*2048 f32
  float* YF  = QF;                        // attention output f32 (alias)
  u16*   YB  = (u16*)KF;                  // attention output bf16 (alias)

  // 1. lambda
  k_lambda<<<1, 256, 0, stream>>>(lq1, lk1, lq2, lk2, LAM);

  // 2. convert inputs to bf16
  k_cvt_bf16<<<8192, 256, 0, stream>>>(x,  XB,  2097152);
  k_cvt_bf16<<<4096, 256, 0, stream>>>(wq, WQB, 1048576);
  k_cvt_bf16<<<4096, 256, 0, stream>>>(wk, WKB, 1048576);
  k_cvt_bf16<<<4096, 256, 0, stream>>>(wv, WVB, 1048576);
  k_cvt_bf16<<<4096, 256, 0, stream>>>(wo, WOB, 1048576);

  // 3. QKV projections (block tile 128x128)
  dim3 gg(16, 32);
  k_gemm_bf16<<<gg, 256, 0, stream>>>(XB, WQB, QF, 1.0f);
  k_gemm_bf16<<<gg, 256, 0, stream>>>(XB, WKB, KF, 1.0f);
  k_gemm_bf16<<<gg, 256, 0, stream>>>(XB, WVB, VF, 1.0f);

  // 4. RMS norm per 128-chunk; V transpose
  k_rmsnorm128<<<8192, 256, 0, stream>>>(QF, QN);
  k_rmsnorm128<<<8192, 256, 0, stream>>>(KF, KN);
  k_transpose_v<<<32768, 256, 0, stream>>>(VF, VT);

  // 5. attention (two-pass softmax, combined differential probs)
  k_attn_sumexp<<<512, 256, 0, stream>>>(QN, KN, LS);
  k_attn_out<<<512, 256, 0, stream>>>(QN, KN, VT, LS, LAM, YF);

  // 6. output projection * (1 - lambda_init)
  k_cvt_bf16<<<8192, 256, 0, stream>>>(YF, YB, 2097152);
  k_gemm_bf16<<<gg, 256, 0, stream>>>(YB, WOB, (float*)d_out, 0.2f);
}
